// RetrievalPrototypeTailNet_73607149519538
// MI455X (gfx1250) — compile-verified, matrix operand order checked by reading, not run on hardware
//
#include <hip/hip_runtime.h>

// ---------- WMMA vector types (CDNA5 gfx1250, wave32) ----------
typedef __attribute__((ext_vector_type(16))) __bf16 v16bf;
typedef __attribute__((ext_vector_type(8)))  float  v8f;
typedef __attribute__((ext_vector_type(4)))  int    v4i;

#define H_DIM 512
#define B_DIM 512
#define L_DIM 64
#define C_CAND 1024
#define K_TOP 16
#define N_PROTO 8

// ---------- CDNA5 async global->LDS (guarded; fallback = ld/st staging) ----
#if defined(__has_builtin)
#if __has_builtin(__builtin_amdgcn_global_load_async_to_lds_b128)
#define HAVE_ASYNC_LDS 1
#endif
#endif

#ifdef HAVE_ASYNC_LDS
__device__ __forceinline__ void async_cp16(const __bf16* g, __bf16* l)
{
    // b128 async DMA: one int4 per lane, global(AS1) -> LDS(AS3), no VGPR data
    __builtin_amdgcn_global_load_async_to_lds_b128(
        (__attribute__((address_space(1))) v4i*)g,
        (__attribute__((address_space(3))) v4i*)l, 0, 0);
}
#if __has_builtin(__builtin_amdgcn_s_wait_asynccnt)
#define WAIT_ASYNC0() __builtin_amdgcn_s_wait_asynccnt(0)
#else
#define WAIT_ASYNC0() asm volatile("s_wait_asynccnt 0x0" ::: "memory")
#endif
#endif

// ===================================================================
// Encoder: masked mean pool + linear + relu  -> local_hidden (f32 + bf16)
// ===================================================================
__global__ __launch_bounds__(256)
void encoder_kernel(const float* __restrict__ seq, const float* __restrict__ mask,
                    const float* __restrict__ stat,
                    const float* __restrict__ W_seq, const float* __restrict__ W_stat,
                    const float* __restrict__ b_enc,
                    float* __restrict__ local, __bf16* __restrict__ local_bf)
{
    int b = blockIdx.x;
    int tid = threadIdx.x;
    __shared__ float pooled[16];
    __shared__ float sstat[8];
    if (tid < 16) {
        float s = 0.f, ms = 0.f;
        for (int l = 0; l < L_DIM; ++l) {
            float m = mask[b * L_DIM + l];
            s  += seq[(size_t)(b * L_DIM + l) * 16 + tid] * m;
            ms += m;
        }
        pooled[tid] = s / fmaxf(ms, 1e-6f);
    } else if (tid < 24) {
        sstat[tid - 16] = stat[b * 8 + (tid - 16)];
    }
    __syncthreads();
    for (int i = 0; i < 2; ++i) {
        int h = tid + i * 256;
        float v = b_enc[h];
        for (int d = 0; d < 16; ++d) v += pooled[d] * W_seq[d * H_DIM + h];
        for (int d = 0; d < 8;  ++d) v += sstat[d]  * W_stat[d * H_DIM + h];
        v = fmaxf(v, 0.f);
        local[(size_t)b * H_DIM + h] = v;
        local_bf[(size_t)b * H_DIM + h] = (__bf16)v;
    }
}

// ===================================================================
// Transpose-convert weight: src [K,N] f32 row-major -> dst [N,K] bf16
// ===================================================================
__global__ __launch_bounds__(256)
void transpose_bf16_kernel(const float* __restrict__ src, __bf16* __restrict__ dst,
                           int K, int N)
{
    int i = blockIdx.x * 256 + threadIdx.x;
    if (i >= K * N) return;
    int n = i / K, k = i % K;
    dst[i] = (__bf16)src[(size_t)k * N + n];
}

// ===================================================================
// bf16 WMMA GEMM: C[M,N] = act(A[M,K] @ Bt[N,K]^T + bias)
// block: 256 thr = 8 waves; tile 128x64; BK=32; wave tile 32x32 (2x2 frags)
// async double-buffered LDS pipeline when GLOBAL_LOAD_ASYNC_TO_LDS exists.
// act: 0=none, 1=relu, 2=sigmoid
// ===================================================================
#define BM 128
#define BN 64
#define BK 32
#define LDS_S 40   // padded LDS stride in bf16 elems (80B rows, 16B aligned)

__global__ __launch_bounds__(256)
void gemm_bf16_kernel(const __bf16* __restrict__ A, const __bf16* __restrict__ Bt,
                      const float* __restrict__ bias, float* __restrict__ C,
                      int M, int N, int K, int act)
{
#ifdef HAVE_ASYNC_LDS
    __shared__ __bf16 As[2][BM * LDS_S];
    __shared__ __bf16 Bs[2][BN * LDS_S];
#else
    __shared__ __bf16 As[1][BM * LDS_S];
    __shared__ __bf16 Bs[1][BN * LDS_S];
#endif
    const int tid  = threadIdx.x;
    const int lane = tid & 31;
    const int wave = tid >> 5;
    const int bm = blockIdx.y * BM;
    const int bn = blockIdx.x * BN;
    const int wrow = (wave >> 1) * 32;   // 4 wave-rows
    const int wcol = (wave & 1) * 32;    // 2 wave-cols
    const int mlane = lane & 15;
    const int kb = (lane >> 4) * 8;      // per-ISA 16-bit A layout: K 0..7 / 8..15

    v8f acc[2][2] = {};

#ifdef HAVE_ASYNC_LDS
    // ---- async double-buffered pipeline: DMA next tile while WMMAs run ----
    auto issue_tile = [&](int k0, int sel) {
        #pragma unroll
        for (int i = 0; i < 2; ++i) {
            int c = tid + i * 256;                   // 512 chunks of 8 bf16 for A
            int row = c >> 2, cc = (c & 3) * 8;
            async_cp16(A + (size_t)(bm + row) * K + k0 + cc,
                       &As[sel][row * LDS_S + cc]);
        }
        {
            int c = tid;                             // 256 chunks for B
            int row = c >> 2, cc = (c & 3) * 8;
            async_cp16(Bt + (size_t)(bn + row) * K + k0 + cc,
                       &Bs[sel][row * LDS_S + cc]);
        }
    };

    const int nT = K / BK;
    issue_tile(0, 0);
    for (int t = 0; t < nT; ++t) {
        const int sel = t & 1;
        WAIT_ASYNC0();                 // our async writes for buf[sel] done
        __syncthreads();               // everyone's writes visible

        v16bf af[2], bf[2];
        #pragma unroll
        for (int f = 0; f < 2; ++f) {
            const __bf16* ap = &As[sel][(wrow + f * 16 + mlane) * LDS_S];
            #pragma unroll
            for (int j = 0; j < 8; ++j) { af[f][j] = ap[kb + j]; af[f][8 + j] = ap[16 + kb + j]; }
            const __bf16* bp = &Bs[sel][(wcol + f * 16 + mlane) * LDS_S];
            #pragma unroll
            for (int j = 0; j < 8; ++j) { bf[f][j] = bp[kb + j]; bf[f][8 + j] = bp[16 + kb + j]; }
        }
        __syncthreads();               // all waves done reading LDS for this tile

        if (t + 1 < nT) issue_tile((t + 1) * BK, sel ^ 1);   // overlaps with WMMA below

        #pragma unroll
        for (int mi = 0; mi < 2; ++mi)
            #pragma unroll
            for (int ni = 0; ni < 2; ++ni)
                acc[mi][ni] = __builtin_amdgcn_wmma_f32_16x16x32_bf16(
                    false, af[mi], false, bf[ni], (short)0, acc[mi][ni], false, false);
    }
#else
    // ---- fallback: ld->VGPR->ds_store staging, single buffer ----
    for (int k0 = 0; k0 < K; k0 += BK) {
        #pragma unroll
        for (int i = 0; i < 4; ++i) {
            int c = tid + i * 256;                  // 1024 chunks of 4 bf16 for A
            int row = c >> 3, cc = (c & 7) * 4;
            *(uint2*)(&As[0][row * LDS_S + cc]) =
                *(const uint2*)(A + (size_t)(bm + row) * K + k0 + cc);
        }
        #pragma unroll
        for (int i = 0; i < 2; ++i) {
            int c = tid + i * 256;                  // 512 chunks for B
            int row = c >> 3, cc = (c & 7) * 4;
            *(uint2*)(&Bs[0][row * LDS_S + cc]) =
                *(const uint2*)(Bt + (size_t)(bn + row) * K + k0 + cc);
        }
        __syncthreads();

        v16bf af[2], bf[2];
        #pragma unroll
        for (int f = 0; f < 2; ++f) {
            const __bf16* ap = &As[0][(wrow + f * 16 + mlane) * LDS_S];
            #pragma unroll
            for (int j = 0; j < 8; ++j) { af[f][j] = ap[kb + j]; af[f][8 + j] = ap[16 + kb + j]; }
            const __bf16* bp = &Bs[0][(wcol + f * 16 + mlane) * LDS_S];
            #pragma unroll
            for (int j = 0; j < 8; ++j) { bf[f][j] = bp[kb + j]; bf[f][8 + j] = bp[16 + kb + j]; }
        }
        #pragma unroll
        for (int mi = 0; mi < 2; ++mi)
            #pragma unroll
            for (int ni = 0; ni < 2; ++ni)
                acc[mi][ni] = __builtin_amdgcn_wmma_f32_16x16x32_bf16(
                    false, af[mi], false, bf[ni], (short)0, acc[mi][ni], false, false);
        __syncthreads();
    }
#endif

    // epilogue: C/D layout -> row = r + 8*(lane>>4), col = lane&15
    #pragma unroll
    for (int mi = 0; mi < 2; ++mi)
        #pragma unroll
        for (int ni = 0; ni < 2; ++ni) {
            int col = bn + wcol + ni * 16 + mlane;
            float bv = bias ? bias[col] : 0.f;
            #pragma unroll
            for (int r = 0; r < 8; ++r) {
                int row = bm + wrow + mi * 16 + r + 8 * (lane >> 4);
                float v = acc[mi][ni][r] + bv;
                if (act == 1) v = fmaxf(v, 0.f);
                else if (act == 2) v = 1.f / (1.f + __expf(-v));
                C[(size_t)row * N + col] = v;
            }
        }
}

// ===================================================================
// Row-wise L2 normalize (in place): x / max(||x||, 1e-12)
// ===================================================================
__global__ __launch_bounds__(256)
void l2norm_rows_kernel(float* __restrict__ x)
{
    int b = blockIdx.x;
    float* row = x + (size_t)b * H_DIM;
    float s = 0.f;
    for (int c = threadIdx.x; c < H_DIM; c += 256) { float v = row[c]; s += v * v; }
    __shared__ float red[256];
    red[threadIdx.x] = s; __syncthreads();
    for (int off = 128; off > 0; off >>= 1) {
        if (threadIdx.x < off) red[threadIdx.x] += red[threadIdx.x + off];
        __syncthreads();
    }
    float inv = 1.f / fmaxf(sqrtf(red[0]), 1e-12f);
    for (int c = threadIdx.x; c < H_DIM; c += 256) row[c] *= inv;
}

// ===================================================================
// sim[b,c] = dot(bank_keys[cand[b,c]], query[b]) ; one wave per pair
// (gather-GEMV: L2-resident bank, WMMA not applicable)
// ===================================================================
__global__ __launch_bounds__(256)
void sim_kernel(const float* __restrict__ query, const float* __restrict__ bank_keys,
                const int* __restrict__ cand, float* __restrict__ sim)
{
    int pair = blockIdx.x * 8 + (threadIdx.x >> 5);
    int lane = threadIdx.x & 31;
    int b = pair >> 10;           // C_CAND = 1024
    int idx = cand[pair];
    const float4* kp = (const float4*)(bank_keys + (size_t)idx * H_DIM);
    const float4* qp = (const float4*)(query + (size_t)b * H_DIM);
    // hint the gather of the next candidate row into cache
    if (pair + 8 < B_DIM * C_CAND) {
        int nidx = cand[pair + 8];
        __builtin_prefetch(bank_keys + (size_t)nidx * H_DIM + lane * 16, 0, 1);
    }
    float s = 0.f;
    #pragma unroll
    for (int i = 0; i < 4; ++i) {
        float4 kv = kp[lane + i * 32];
        float4 qv = qp[lane + i * 32];
        s += kv.x * qv.x + kv.y * qv.y + kv.z * qv.z + kv.w * qv.w;
    }
    #pragma unroll
    for (int off = 16; off > 0; off >>= 1) s += __shfl_xor(s, off, 32);
    if (lane == 0) sim[pair] = s;
}

// ===================================================================
// top-16 of 1024 + softmax(T=0.2) + weighted gather of bank_values
// ===================================================================
__global__ __launch_bounds__(256)
void topk_donor_kernel(const float* __restrict__ sim, const int* __restrict__ cand,
                       const float* __restrict__ bank_values, float* __restrict__ donor)
{
    int b = blockIdx.x;
    int tid = threadIdx.x;
    __shared__ float sv[C_CAND];
    __shared__ float rmax[256];
    __shared__ int   rpos[256];
    __shared__ float topw[K_TOP];
    __shared__ int   topi[K_TOP];
    for (int i = tid; i < C_CAND; i += 256) sv[i] = sim[(size_t)b * C_CAND + i];
    __syncthreads();
    for (int k = 0; k < K_TOP; ++k) {
        float best = -3.402823466e38f; int bp = 0;
        for (int i = tid; i < C_CAND; i += 256) {
            float v = sv[i];
            if (v > best) { best = v; bp = i; }
        }
        rmax[tid] = best; rpos[tid] = bp; __syncthreads();
        for (int off = 128; off > 0; off >>= 1) {
            if (tid < off && rmax[tid + off] > rmax[tid]) {
                rmax[tid] = rmax[tid + off]; rpos[tid] = rpos[tid + off];
            }
            __syncthreads();
        }
        if (tid == 0) { topw[k] = rmax[0]; topi[k] = rpos[0]; sv[rpos[0]] = -3.402823466e38f; }
        __syncthreads();
    }
    if (tid == 0) {
        float mx = topw[0];
        for (int k = 1; k < K_TOP; ++k) mx = fmaxf(mx, topw[k]);
        float sum = 0.f;
        for (int k = 0; k < K_TOP; ++k) { float e = __expf((topw[k] - mx) * 5.0f); topw[k] = e; sum += e; }
        float inv = 1.f / sum;
        for (int k = 0; k < K_TOP; ++k) topw[k] *= inv;
    }
    __syncthreads();
    for (int h = tid; h < H_DIM; h += 256) {
        float acc = 0.f;
        #pragma unroll
        for (int k = 0; k < K_TOP; ++k) {
            int row = cand[(size_t)b * C_CAND + topi[k]];
            acc += topw[k] * bank_values[(size_t)row * H_DIM + h];
        }
        donor[(size_t)b * H_DIM + h] = acc;
    }
}

// ===================================================================
// prototype attention: pw = softmax(q . l2norm(proto)); out = pw @ proto
// ===================================================================
__global__ __launch_bounds__(256)
void proto_kernel(const float* __restrict__ query, const float* __restrict__ protos,
                  float* __restrict__ proto_hidden)
{
    int b = blockIdx.x;
    int wave = threadIdx.x >> 5, lane = threadIdx.x & 31;
    __shared__ float pw[N_PROTO];
    {
        const float* pr = protos + wave * H_DIM;
        const float* q  = query + (size_t)b * H_DIM;
        float dq = 0.f, dp = 0.f;
        for (int i = lane; i < H_DIM; i += 32) { float pv = pr[i]; dq += q[i] * pv; dp += pv * pv; }
        #pragma unroll
        for (int off = 16; off > 0; off >>= 1) {
            dq += __shfl_xor(dq, off, 32);
            dp += __shfl_xor(dp, off, 32);
        }
        if (lane == 0) pw[wave] = dq / fmaxf(sqrtf(dp), 1e-12f);
    }
    __syncthreads();
    if (threadIdx.x == 0) {
        float mx = pw[0];
        for (int p = 1; p < N_PROTO; ++p) mx = fmaxf(mx, pw[p]);
        float sum = 0.f;
        for (int p = 0; p < N_PROTO; ++p) { float e = __expf(pw[p] - mx); pw[p] = e; sum += e; }
        for (int p = 0; p < N_PROTO; ++p) pw[p] /= sum;
    }
    __syncthreads();
    for (int h = threadIdx.x; h < H_DIM; h += 256) {
        float acc = 0.f;
        #pragma unroll
        for (int p = 0; p < N_PROTO; ++p) acc += pw[p] * protos[p * H_DIM + h];
        proto_hidden[(size_t)b * H_DIM + h] = acc;
    }
}

// ===================================================================
// pack bf16 concat activations for the transfer/gate GEMMs
// ===================================================================
__global__ __launch_bounds__(256)
void pack_kernel(const float* __restrict__ local, const float* __restrict__ donor,
                 const float* __restrict__ proto,
                 __bf16* __restrict__ A_dp, __bf16* __restrict__ A_ldp)
{
    int i = blockIdx.x * 256 + threadIdx.x;   // over B*H
    if (i >= B_DIM * H_DIM) return;
    int b = i >> 9, h = i & 511;
    float lv = local[i], dv = donor[i], pv = proto[i];
    A_dp[(size_t)b * 1024 + h]         = (__bf16)dv;
    A_dp[(size_t)b * 1024 + 512 + h]   = (__bf16)pv;
    A_ldp[(size_t)b * 1536 + h]        = (__bf16)lv;
    A_ldp[(size_t)b * 1536 + 512 + h]  = (__bf16)dv;
    A_ldp[(size_t)b * 1536 + 1024 + h] = (__bf16)pv;
}

// ===================================================================
// fused_pre = gate*local + (1-gate)*transfer  (bf16 for final GEMM)
// ===================================================================
__global__ __launch_bounds__(256)
void fuse_kernel(const float* __restrict__ gate, const float* __restrict__ local,
                 const float* __restrict__ transfer, __bf16* __restrict__ A_f)
{
    int i = blockIdx.x * 256 + threadIdx.x;
    if (i >= B_DIM * H_DIM) return;
    float g = gate[i];
    A_f[i] = (__bf16)(g * local[i] + (1.f - g) * transfer[i]);
}

// ===================================================================
// heads: 6 quantile dots (sorted per target) + 2 event dots
// out[b, 0..5] = sorted quantiles, out[b, 6..7] = logits
// ===================================================================
__global__ __launch_bounds__(256)
void heads_kernel(const float* __restrict__ fused, const float* __restrict__ W_quant,
                  const float* __restrict__ b_quant, const float* __restrict__ W_ev,
                  const float* __restrict__ b_ev, float* __restrict__ out)
{
    int b = blockIdx.x;
    int wave = threadIdx.x >> 5, lane = threadIdx.x & 31;
    __shared__ float vals[8];
    const float* f = fused + (size_t)b * H_DIM;
    float acc = 0.f;
    if (wave < 6) {
        int t = wave / 3, q = wave % 3;
        const float* w = W_quant + (size_t)t * H_DIM * 3 + q;   // [T,H,Q]
        for (int h = lane; h < H_DIM; h += 32) acc += f[h] * w[h * 3];
    } else {
        int t = wave - 6;
        const float* w = W_ev + t * H_DIM;
        for (int h = lane; h < H_DIM; h += 32) acc += f[h] * w[h];
    }
    #pragma unroll
    for (int off = 16; off > 0; off >>= 1) acc += __shfl_xor(acc, off, 32);
    if (lane == 0) {
        if (wave < 6) vals[wave] = acc + b_quant[(wave / 3) * 3 + (wave % 3)];
        else          vals[wave] = acc + b_ev[wave - 6];
    }
    __syncthreads();
    if (threadIdx.x == 0) {
        float* o = out + (size_t)b * 8;
        for (int t = 0; t < 2; ++t) {
            float a0 = vals[t * 3 + 0], a1 = vals[t * 3 + 1], a2 = vals[t * 3 + 2];
            float lo = fminf(a0, fminf(a1, a2));
            float hi = fmaxf(a0, fmaxf(a1, a2));
            float mid = a0 + a1 + a2 - lo - hi;
            o[t * 3 + 0] = lo; o[t * 3 + 1] = mid; o[t * 3 + 2] = hi;
        }
        o[6] = vals[6]; o[7] = vals[7];
    }
}

// ===================================================================
// launch
// ===================================================================
extern "C" void kernel_launch(void* const* d_in, const int* in_sizes, int n_in,
                              void* d_out, int out_size, void* d_ws, size_t ws_size,
                              hipStream_t stream)
{
    const float* seq        = (const float*)d_in[0];
    const float* mask       = (const float*)d_in[1];
    const float* stat       = (const float*)d_in[2];
    const float* bank_keys  = (const float*)d_in[3];
    const float* bank_vals  = (const float*)d_in[4];
    const int*   cand       = (const int*)d_in[5];
    const float* W_seq      = (const float*)d_in[6];
    const float* W_stat     = (const float*)d_in[7];
    const float* b_enc      = (const float*)d_in[8];
    const float* W_q        = (const float*)d_in[9];
    const float* b_q        = (const float*)d_in[10];
    const float* protos     = (const float*)d_in[11];
    const float* W_tr       = (const float*)d_in[12];
    const float* b_tr       = (const float*)d_in[13];
    const float* W_gate     = (const float*)d_in[14];
    const float* b_gate     = (const float*)d_in[15];
    const float* W_out      = (const float*)d_in[16];
    const float* b_out      = (const float*)d_in[17];
    const float* W_quant    = (const float*)d_in[18];
    const float* b_quant    = (const float*)d_in[19];
    const float* W_ev       = (const float*)d_in[20];
    const float* b_ev       = (const float*)d_in[21];
    float* out = (float*)d_out;

    char* ws = (char*)d_ws;
    auto alloc = [&](size_t bytes) -> char* {
        char* p = ws;
        ws += (bytes + 255) & ~(size_t)255;
        return p;
    };
    float*  local     = (float*)alloc((size_t)B_DIM * H_DIM * 4);
    float*  query     = (float*)alloc((size_t)B_DIM * H_DIM * 4);
    float*  sim       = (float*)alloc((size_t)B_DIM * C_CAND * 4);
    float*  donor     = (float*)alloc((size_t)B_DIM * H_DIM * 4);
    float*  protoH    = (float*)alloc((size_t)B_DIM * H_DIM * 4);
    float*  transfer  = (float*)alloc((size_t)B_DIM * H_DIM * 4);
    float*  gate      = (float*)alloc((size_t)B_DIM * H_DIM * 4);
    float*  fused     = (float*)alloc((size_t)B_DIM * H_DIM * 4);
    __bf16* local_bf  = (__bf16*)alloc((size_t)B_DIM * H_DIM * 2);
    __bf16* A_dp      = (__bf16*)alloc((size_t)B_DIM * 1024 * 2);
    __bf16* A_ldp     = (__bf16*)alloc((size_t)B_DIM * 1536 * 2);
    __bf16* A_f       = (__bf16*)alloc((size_t)B_DIM * H_DIM * 2);
    __bf16* Wq_t      = (__bf16*)alloc((size_t)512 * 512 * 2);
    __bf16* Wtr_t     = (__bf16*)alloc((size_t)512 * 1024 * 2);
    __bf16* Wgate_t   = (__bf16*)alloc((size_t)512 * 1536 * 2);
    __bf16* Wout_t    = (__bf16*)alloc((size_t)512 * 512 * 2);

    // weight transposes (f32 -> bf16 [N,K])
    transpose_bf16_kernel<<<(512 * 512 + 255) / 256, 256, 0, stream>>>(W_q, Wq_t, 512, 512);
    transpose_bf16_kernel<<<(1024 * 512 + 255) / 256, 256, 0, stream>>>(W_tr, Wtr_t, 1024, 512);
    transpose_bf16_kernel<<<(1536 * 512 + 255) / 256, 256, 0, stream>>>(W_gate, Wgate_t, 1536, 512);
    transpose_bf16_kernel<<<(512 * 512 + 255) / 256, 256, 0, stream>>>(W_out, Wout_t, 512, 512);

    // encoder
    encoder_kernel<<<B_DIM, 256, 0, stream>>>(seq, mask, stat, W_seq, W_stat, b_enc,
                                              local, local_bf);

    // query = l2norm(local @ W_q + b_q)   — WMMA GEMM
    dim3 ggrid(H_DIM / BN, B_DIM / BM);
    gemm_bf16_kernel<<<ggrid, 256, 0, stream>>>(local_bf, Wq_t, b_q, query,
                                                B_DIM, H_DIM, 512, 0);
    l2norm_rows_kernel<<<B_DIM, 256, 0, stream>>>(query);

    // gather-dot sim
    sim_kernel<<<(B_DIM * C_CAND) / 8, 256, 0, stream>>>(query, bank_keys, cand, sim);

    // top-k + softmax + donor
    topk_donor_kernel<<<B_DIM, 256, 0, stream>>>(sim, cand, bank_vals, donor);

    // prototype attention
    proto_kernel<<<B_DIM, 256, 0, stream>>>(query, protos, protoH);

    // pack concat activations
    pack_kernel<<<(B_DIM * H_DIM + 255) / 256, 256, 0, stream>>>(local, donor, protoH,
                                                                 A_dp, A_ldp);

    // transfer = relu([donor,proto] @ W_tr + b_tr)
    gemm_bf16_kernel<<<ggrid, 256, 0, stream>>>(A_dp, Wtr_t, b_tr, transfer,
                                                B_DIM, H_DIM, 1024, 1);
    // gate = sigmoid([local,donor,proto] @ W_gate + b_gate)
    gemm_bf16_kernel<<<ggrid, 256, 0, stream>>>(A_ldp, Wgate_t, b_gate, gate,
                                                B_DIM, H_DIM, 1536, 2);

    // fuse
    fuse_kernel<<<(B_DIM * H_DIM + 255) / 256, 256, 0, stream>>>(gate, local, transfer, A_f);

    // fused = relu(fused_pre @ W_out + b_out)
    gemm_bf16_kernel<<<ggrid, 256, 0, stream>>>(A_f, Wout_t, b_out, fused,
                                                B_DIM, H_DIM, 512, 1);

    // heads
    heads_kernel<<<B_DIM, 256, 0, stream>>>(fused, W_quant, b_quant, W_ev, b_ev, out);
}
